// _Encoder_11355893531023
// MI455X (gfx1250) — compile-verified, matrix operand order checked by reading, not run on hardware
//
#include <hip/hip_runtime.h>
#include <hip/hip_bf16.h>
#include <cstdint>

typedef __attribute__((ext_vector_type(16))) __bf16 v16bf;
typedef __attribute__((ext_vector_type(8)))  float  v8f;

#define D_MODEL 1024
#define NHEAD 16
#define HD 64
#define SEQ 2048
#define BATCH 2
#define ROWS (BATCH * SEQ)  // 4096

union FragBF {
  uint4 u[2];
  v16bf v;
};

__device__ __forceinline__ v8f vzero8() {
  v8f z = {0.f, 0.f, 0.f, 0.f, 0.f, 0.f, 0.f, 0.f};
  return z;
}

// ---- CDNA5 async global->LDS copy (ASYNCcnt path) ----
__device__ __forceinline__ void async_copy16(void* lds_dst, const void* gsrc) {
  unsigned int l = (unsigned int)(uintptr_t)lds_dst;  // low 32 bits = LDS offset
  asm volatile("global_load_async_to_lds_b128 %0, %1, off"
               :: "v"(l), "v"(gsrc)
               : "memory");
}
__device__ __forceinline__ void async_wait0() {
  asm volatile("s_wait_asynccnt 0" ::: "memory");
}

// =====================================================================
// bf16 GEMM: C[M][N] = A[M][K] * B[K][N], B given transposed (BT[N][K]).
// Block tile 256(M) x 128(N), BK=64, 8 waves (4x2), wave tile 64x64.
// 16 WMMAs per 16 ds_load_b128 per 32-K half-step; double-buffered async LDS.
// EPI: 0 = bf16 out (+bias), 1 = bf16 out (+bias, exact GELU), 2 = f32 out (+bias)
// =====================================================================
template<int EPI>
__global__ __launch_bounds__(256)
void gemm_bf16_k(const __bf16* __restrict__ A, int lda,
                 const __bf16* __restrict__ BT, int ldb,
                 const float* __restrict__ bias,
                 void* __restrict__ Cout, int ldc,
                 int M, int N, int K) {
  __shared__ __bf16 sA[2][256][72];   // 256 rows x 64 K (+8 pad)
  __shared__ __bf16 sB[2][128][72];   // 128 cols x 64 K (+8 pad)
  const int t    = threadIdx.x;
  const int lane = t & 31;
  const int w    = t >> 5;
  const int wm   = w & 3;        // 0..3 -> 64-row slab
  const int wn   = w >> 2;       // 0..1 -> 64-col slab
  const int lr   = lane & 15;
  const int half = lane >> 4;
  const int klo  = half * 8;
  const int m0   = blockIdx.y * 256;
  const int n0   = blockIdx.x * 128;

  v8f acc[4][4];
  #pragma unroll
  for (int i = 0; i < 4; ++i)
    #pragma unroll
    for (int j = 0; j < 4; ++j)
      acc[i][j] = vzero8();

  // Stage one 256x64 A tile + 128x64 B tile: 3072 16B chunks, 12 per thread.
  auto load_tiles = [&](int s, int kt) {
    const int k0 = kt * 64;
    #pragma unroll
    for (int i = 0; i < 12; ++i) {
      int c = t + i * 256;
      if (c < 2048) {               // uniform per i (i < 8)
        int row = c >> 3;
        int kc  = (c & 7) * 8;
        async_copy16(&sA[s][row][kc], A + (size_t)(m0 + row) * lda + k0 + kc);
      } else {
        int c2  = c - 2048;
        int row = c2 >> 3;
        int kc  = (c2 & 7) * 8;
        async_copy16(&sB[s][row][kc], BT + (size_t)(n0 + row) * ldb + k0 + kc);
      }
    }
  };

  const int nk = K >> 6;
  load_tiles(0, 0);
  async_wait0();
  __syncthreads();

  int cur = 0;
  for (int kt = 0; kt < nk; ++kt) {
    if (kt + 1 < nk) load_tiles(cur ^ 1, kt + 1);

    #pragma unroll
    for (int kk = 0; kk < 2; ++kk) {
      FragBF b[4];
      #pragma unroll
      for (int j = 0; j < 4; ++j) {
        const __bf16* p = &sB[cur][wn * 64 + j * 16 + lr][kk * 32 + klo];
        b[j].u[0] = *(const uint4*)p;
        b[j].u[1] = *(const uint4*)(p + 16);
      }
      #pragma unroll
      for (int i = 0; i < 4; ++i) {
        FragBF a;
        const __bf16* p = &sA[cur][wm * 64 + i * 16 + lr][kk * 32 + klo];
        a.u[0] = *(const uint4*)p;
        a.u[1] = *(const uint4*)(p + 16);
        #pragma unroll
        for (int j = 0; j < 4; ++j)
          acc[i][j] = __builtin_amdgcn_wmma_f32_16x16x32_bf16(
              false, a.v, false, b[j].v, (short)0, acc[i][j], false, false);
      }
    }

    async_wait0();
    __syncthreads();
    cur ^= 1;
  }

  #pragma unroll
  for (int i = 0; i < 4; ++i) {
    #pragma unroll
    for (int j = 0; j < 4; ++j) {
      int col  = n0 + wn * 64 + j * 16 + lr;
      float bv = bias ? bias[col] : 0.0f;
      #pragma unroll
      for (int r = 0; r < 8; ++r) {
        int row = m0 + wm * 64 + i * 16 + half * 8 + r;
        float v = acc[i][j][r] + bv;
        if (EPI == 1) v = 0.5f * v * (1.0f + erff(v * 0.70710678118654752f));
        if (EPI <= 1) ((__bf16*)Cout)[(size_t)row * ldc + col] = (__bf16)v;
        else          ((float*) Cout)[(size_t)row * ldc + col] = v;
      }
    }
  }
}

// =====================================================================
// Flash attention: one block = 128 query rows of one (b,h); 8 waves x 16 rows.
// Key loop: 32 keys/tile. Online softmax per row, done inside 16-lane halves.
// =====================================================================
__global__ __launch_bounds__(256)
void attn_k(const __bf16* __restrict__ qkv, __bf16* __restrict__ outp) {
  __shared__ __bf16 sK[32][72];       // keys x 64 head-cols (+pad)
  __shared__ __bf16 sVT[64][40];      // head-cols x 32 keys (+pad)
  __shared__ __bf16 sP[8][16][40];    // per-wave P tile 16x32 (+pad)

  const int t    = threadIdx.x;
  const int lane = t & 31;
  const int w    = t >> 5;
  const int lr   = lane & 15;
  const int half = lane >> 4;
  const int klo  = half * 8;
  const int h    = blockIdx.y;
  const int b    = blockIdx.z;
  const size_t rowbase = (size_t)b * SEQ;
  const int q0   = blockIdx.x * 128 + w * 16;

  // Q fragments (held in registers for the whole kernel); K-dim = 64 -> 2 frags.
  FragBF aq[2];
  {
    const __bf16* qp = qkv + (rowbase + q0 + lr) * 3072 + h * HD;
    #pragma unroll
    for (int kk = 0; kk < 2; ++kk) {
      aq[kk].u[0] = *(const uint4*)(qp + kk * 32 + klo);
      aq[kk].u[1] = *(const uint4*)(qp + kk * 32 + klo + 16);
    }
  }

  v8f o[4];
  #pragma unroll
  for (int j2 = 0; j2 < 4; ++j2) o[j2] = vzero8();
  float mrow[8], lrow[8];
  #pragma unroll
  for (int r = 0; r < 8; ++r) { mrow[r] = -1e30f; lrow[r] = 0.0f; }

  const int vrow_l = t >> 3;        // 0..31 (key within tile)
  const int vcol_l = (t & 7) * 8;   // 0..56 (head col chunk)

  for (int kt = 0; kt < SEQ / 32; ++kt) {
    __syncthreads();  // previous tile's readers done before overwrite
    const int key0 = kt * 32;
    // Stage K tile (async DMA to LDS), rows = keys (this IS B^T for Q@K^T)
    async_copy16(&sK[vrow_l][vcol_l],
                 qkv + (rowbase + key0 + vrow_l) * 3072 + D_MODEL + h * HD + vcol_l);
    // Stage V tile transposed (cols x keys) so P@V B-frags are contiguous
    {
      const __bf16* vp =
          qkv + (rowbase + key0 + vrow_l) * 3072 + 2 * D_MODEL + h * HD + vcol_l;
      uint4 d = *(const uint4*)vp;
      __bf16 tmp[8];
      *(uint4*)tmp = d;
      #pragma unroll
      for (int e = 0; e < 8; ++e) sVT[vcol_l + e][vrow_l] = tmp[e];
    }
    async_wait0();
    __syncthreads();

    // S = (Q K^T) * 1/sqrt(hd) for 16 rows x 32 keys
    v8f s[2];
    #pragma unroll
    for (int j = 0; j < 2; ++j) {
      v8f sv = vzero8();
      #pragma unroll
      for (int kk = 0; kk < 2; ++kk) {
        FragBF bk;
        const __bf16* p = &sK[j * 16 + lr][kk * 32 + klo];
        bk.u[0] = *(const uint4*)p;
        bk.u[1] = *(const uint4*)(p + 16);
        sv = __builtin_amdgcn_wmma_f32_16x16x32_bf16(
            false, aq[kk].v, false, bk.v, (short)0, sv, false, false);
      }
      s[j] = sv;
    }

    // Online softmax: each output row lives in one 16-lane half (VGPR r <-> row)
    #pragma unroll
    for (int r = 0; r < 8; ++r) {
      float s0 = s[0][r] * 0.125f;
      float s1 = s[1][r] * 0.125f;
      float mx = fmaxf(s0, s1);
      mx = fmaxf(mx, __shfl_xor(mx, 1));
      mx = fmaxf(mx, __shfl_xor(mx, 2));
      mx = fmaxf(mx, __shfl_xor(mx, 4));
      mx = fmaxf(mx, __shfl_xor(mx, 8));
      float mnew = fmaxf(mrow[r], mx);
      float p0 = __expf(s0 - mnew);
      float p1 = __expf(s1 - mnew);
      float rs = p0 + p1;
      rs += __shfl_xor(rs, 1);
      rs += __shfl_xor(rs, 2);
      rs += __shfl_xor(rs, 4);
      rs += __shfl_xor(rs, 8);
      float corr = __expf(mrow[r] - mnew);
      lrow[r] = lrow[r] * corr + rs;
      mrow[r] = mnew;
      #pragma unroll
      for (int j2 = 0; j2 < 4; ++j2) o[j2][r] *= corr;
      // Redistribute P from C-layout to A-layout via per-wave LDS patch
      sP[w][half * 8 + r][lr]      = (__bf16)p0;
      sP[w][half * 8 + r][16 + lr] = (__bf16)p1;
    }

    // O += P @ V   (A = P 16x32, B^T = sVT rows)
    FragBF ap;
    const __bf16* pp = &sP[w][lr][klo];
    ap.u[0] = *(const uint4*)pp;
    ap.u[1] = *(const uint4*)(pp + 16);
    #pragma unroll
    for (int j2 = 0; j2 < 4; ++j2) {
      FragBF bv;
      const __bf16* vp2 = &sVT[j2 * 16 + lr][klo];
      bv.u[0] = *(const uint4*)vp2;
      bv.u[1] = *(const uint4*)(vp2 + 16);
      o[j2] = __builtin_amdgcn_wmma_f32_16x16x32_bf16(
          false, ap.v, false, bv.v, (short)0, o[j2], false, false);
    }
  }

  // Normalize by row sums and store bf16
  #pragma unroll
  for (int j2 = 0; j2 < 4; ++j2) {
    int col = h * HD + j2 * 16 + lr;
    #pragma unroll
    for (int r = 0; r < 8; ++r) {
      size_t row = rowbase + q0 + half * 8 + r;
      outp[row * D_MODEL + col] = (__bf16)(o[j2][r] / lrow[r]);
    }
  }
}

// =====================================================================
// Fused residual-add + LayerNorm (unbiased var, ddof=1), dual f32/bf16 out
// =====================================================================
__global__ __launch_bounds__(256)
void ln_k(const float* __restrict__ src, const float* __restrict__ resid,
          const float* __restrict__ gamma, const float* __restrict__ beta,
          float* __restrict__ outf, __bf16* __restrict__ outbf) {
  __shared__ float sbuf[256];
  const int t = threadIdx.x;
  const size_t base = (size_t)blockIdx.x * D_MODEL;
  float v[4];
  #pragma unroll
  for (int e = 0; e < 4; ++e) {
    int c = t + e * 256;
    v[e] = src[base + c] + resid[base + c];
  }
  sbuf[t] = v[0] + v[1] + v[2] + v[3];
  __syncthreads();
  for (int off = 128; off > 0; off >>= 1) {
    if (t < off) sbuf[t] += sbuf[t + off];
    __syncthreads();
  }
  float mean = sbuf[0] * (1.0f / D_MODEL);
  __syncthreads();
  float sq = 0.f;
  #pragma unroll
  for (int e = 0; e < 4; ++e) { float d = v[e] - mean; sq += d * d; }
  sbuf[t] = sq;
  __syncthreads();
  for (int off = 128; off > 0; off >>= 1) {
    if (t < off) sbuf[t] += sbuf[t + off];
    __syncthreads();
  }
  float inv = rsqrtf(sbuf[0] * (1.0f / (D_MODEL - 1)) + 1e-5f);
  #pragma unroll
  for (int e = 0; e < 4; ++e) {
    int c = t + e * 256;
    float y = (v[e] - mean) * inv * gamma[c] + beta[c];
    outf[base + c] = y;
    if (outbf) outbf[base + c] = (__bf16)y;
  }
}

// =====================================================================
// f32 -> bf16 convert, and f32 [K][N] -> bf16 transpose [N][K]
// =====================================================================
__global__ __launch_bounds__(256)
void conv_bf16_k(const float* __restrict__ x, __bf16* __restrict__ o, int n) {
  int i = blockIdx.x * 256 + threadIdx.x;
  if (i < n) o[i] = (__bf16)x[i];
}

__global__ __launch_bounds__(256)
void transpose_bf16_k(const float* __restrict__ W, __bf16* __restrict__ WT,
                      int K, int N) {  // W[K][N] -> WT[N][K]
  __shared__ float tile[32][33];
  const int t  = threadIdx.x;
  const int tx = t & 31;
  const int ty = t >> 5;  // 0..7
  const int bn = blockIdx.x * 32;
  const int bk = blockIdx.y * 32;
  #pragma unroll
  for (int p = 0; p < 4; ++p)
    tile[ty + p * 8][tx] = W[(size_t)(bk + ty + p * 8) * N + bn + tx];
  __syncthreads();
  #pragma unroll
  for (int p = 0; p < 4; ++p)
    WT[(size_t)(bn + ty + p * 8) * K + bk + tx] = (__bf16)tile[tx][ty + p * 8];
}

// =====================================================================
extern "C" void kernel_launch(void* const* d_in, const int* in_sizes, int n_in,
                              void* d_out, int out_size, void* d_ws, size_t ws_size,
                              hipStream_t stream) {
  (void)in_sizes; (void)n_in; (void)out_size; (void)ws_size;
  const float* x      = (const float*)d_in[0];
  const float* w_qkv  = (const float*)d_in[1];
  const float* b_qkv  = (const float*)d_in[2];
  const float* w_out  = (const float*)d_in[3];
  const float* b_out  = (const float*)d_in[4];
  const float* w_ff1  = (const float*)d_in[5];
  const float* b_ff1  = (const float*)d_in[6];
  const float* w_ff2  = (const float*)d_in[7];
  const float* b_ff2  = (const float*)d_in[8];
  const float* gamma1 = (const float*)d_in[9];
  const float* beta1  = (const float*)d_in[10];
  const float* gamma2 = (const float*)d_in[11];
  const float* beta2  = (const float*)d_in[12];

  char* ws = (char*)d_ws;
  const size_t MB = 1ull << 20;
  __bf16* xb    = (__bf16*)(ws + 0);        //  8 MB  x as bf16
  __bf16* WqkvT = (__bf16*)(ws +   8 * MB); //  6 MB
  __bf16* WoutT = (__bf16*)(ws +  14 * MB); //  2 MB
  __bf16* Wff1T = (__bf16*)(ws +  16 * MB); //  8 MB
  __bf16* Wff2T = (__bf16*)(ws +  24 * MB); //  8 MB
  __bf16* qkvb  = (__bf16*)(ws +  32 * MB); // 24 MB (dead after attention)
  __bf16* ff1b  = (__bf16*)(ws +  32 * MB); // 32 MB (reuses qkv region)
  __bf16* attnb = (__bf16*)(ws +  64 * MB); //  8 MB
  float*  gout  = (float*) (ws +  72 * MB); // 16 MB (proj f32, then ff2 f32)
  float*  h1f   = (float*) (ws +  88 * MB); // 16 MB
  __bf16* h1b   = (__bf16*)(ws + 104 * MB); //  8 MB

  // 1. convert inputs / transpose weights to bf16
  conv_bf16_k<<<(ROWS * D_MODEL) / 256, 256, 0, stream>>>(x, xb, ROWS * D_MODEL);
  transpose_bf16_k<<<dim3(3 * D_MODEL / 32, D_MODEL / 32), 256, 0, stream>>>(
      w_qkv, WqkvT, D_MODEL, 3 * D_MODEL);
  transpose_bf16_k<<<dim3(D_MODEL / 32, D_MODEL / 32), 256, 0, stream>>>(
      w_out, WoutT, D_MODEL, D_MODEL);
  transpose_bf16_k<<<dim3(4 * D_MODEL / 32, D_MODEL / 32), 256, 0, stream>>>(
      w_ff1, Wff1T, D_MODEL, 4 * D_MODEL);
  transpose_bf16_k<<<dim3(D_MODEL / 32, 4 * D_MODEL / 32), 256, 0, stream>>>(
      w_ff2, Wff2T, 4 * D_MODEL, D_MODEL);

  // 2. qkv = x @ Wqkv + b (bf16 out)
  gemm_bf16_k<0><<<dim3(3 * D_MODEL / 128, ROWS / 256), 256, 0, stream>>>(
      xb, D_MODEL, WqkvT, D_MODEL, b_qkv, qkvb, 3 * D_MODEL, ROWS, 3 * D_MODEL, D_MODEL);

  // 3. flash attention
  attn_k<<<dim3(SEQ / 128, NHEAD, BATCH), 256, 0, stream>>>(qkvb, attnb);

  // 4. out projection (f32 out + bias)
  gemm_bf16_k<2><<<dim3(D_MODEL / 128, ROWS / 256), 256, 0, stream>>>(
      attnb, D_MODEL, WoutT, D_MODEL, b_out, gout, D_MODEL, ROWS, D_MODEL, D_MODEL);

  // 5. LN1(proj + x) -> h1 (f32 + bf16)
  ln_k<<<ROWS, 256, 0, stream>>>(gout, x, gamma1, beta1, h1f, h1b);

  // 6. ff1 = GELU(h1 @ Wff1 + b) (bf16 out)
  gemm_bf16_k<1><<<dim3(4 * D_MODEL / 128, ROWS / 256), 256, 0, stream>>>(
      h1b, D_MODEL, Wff1T, D_MODEL, b_ff1, ff1b, 4 * D_MODEL, ROWS, 4 * D_MODEL, D_MODEL);

  // 7. ff2 (f32 out + bias)
  gemm_bf16_k<2><<<dim3(D_MODEL / 128, ROWS / 256), 256, 0, stream>>>(
      ff1b, 4 * D_MODEL, Wff2T, 4 * D_MODEL, b_ff2, gout, D_MODEL, ROWS, D_MODEL, 4 * D_MODEL);

  // 8. LN2(ff2 + h1) -> out (f32)
  ln_k<<<ROWS, 256, 0, stream>>>(gout, h1f, gamma2, beta2, (float*)d_out, (__bf16*)nullptr);
}
